// Decoder_77945066488376
// MI455X (gfx1250) — matrix operand order repaired and checked by reading, not compiled, not run on hardware
//
#include <hip/hip_runtime.h>

// ---------------------------------------------------------------------------
// LSTM decoder for MI455X (gfx1250, wave32, WMMA, async-LDS).
//   B=64, H=1024, T'=127, gates=4096.
//   1) prep: bf16 conversion, bias sum, embedding gather, state init.
//   2) gemm_in: G[8128,4096] = X@W_ih^T + b. bf16 WMMA, f32 accum. 8 waves,
//      WG tile 64x256, A tile double-buffered in LDS via async global->LDS
//      copies (copy of chunk k+1 overlaps WMMA of chunk k).
//   3) lstm: persistent 64 WGs x 4 waves. Per WG: W_hh tile (128KB) resident
//      in LDS; h (128KB) async-staged per step; wave g computes gate g's
//      [64x16] WMMA tile; gates exchanged via LDS; f32 cell update;
//      double-buffered bf16 h; one device-scope barrier per step.
// ---------------------------------------------------------------------------

typedef __attribute__((ext_vector_type(8)))  __bf16 bf16x8;
typedef __attribute__((ext_vector_type(16))) __bf16 bf16x16;
typedef __attribute__((ext_vector_type(8)))  float  f32x8;
typedef __attribute__((ext_vector_type(4)))  int    i32x4;

#define HIDN 1024
#define BATCH 64
#define TPRIME 127
#define GDIM 4096
#define LSTM_WGS 64

#if __has_builtin(__builtin_amdgcn_global_load_async_to_lds_b128) && \
    __has_builtin(__builtin_amdgcn_s_wait_asynccnt)
#define HAVE_ASYNC_LDS 1
#else
#define HAVE_ASYNC_LDS 0
#endif

// Builtin signature (from hipcc diagnostic): (v4i AS1*, v4i AS3*, imm, imm)
#define GPTR(p) ((__attribute__((address_space(1))) i32x4*)(uintptr_t)(p))
#define LPTR(p) ((__attribute__((address_space(3))) i32x4*)(uintptr_t)(p))

__device__ inline void copy_g2l_b128(const unsigned short* g, unsigned short* l) {
#if HAVE_ASYNC_LDS
  __builtin_amdgcn_global_load_async_to_lds_b128(GPTR(g), LPTR(l), 0, 0);
#else
  *(bf16x8*)l = *(const bf16x8*)g;
#endif
}

#if HAVE_ASYNC_LDS
#define ASYNC_WAIT(n) __builtin_amdgcn_s_wait_asynccnt(n)
#else
#define ASYNC_WAIT(n)
#endif

__device__ inline void async_join() {  // wait all + WG barrier
  ASYNC_WAIT(0);
  __syncthreads();
}

__device__ inline unsigned short f2bf(float x) {
  unsigned u = __builtin_bit_cast(unsigned, x);
  unsigned r = u + 0x7FFFu + ((u >> 16) & 1u);  // round-to-nearest-even
  return (unsigned short)(r >> 16);
}

// A fragment (16x32 bf16, MxK). Lane l<16: row=l, K in {0..7,16..23};
// lane>=16: row=l-16, K in {8..15,24..31}.
__device__ inline bf16x16 load_frag_A(const unsigned short* base, int stride, int lane) {
  int half = (lane >> 4) & 1;
  const unsigned short* p = base + (size_t)(lane & 15) * stride + half * 8;
  bf16x8 lo = *(const bf16x8*)(p);
  bf16x8 hi = *(const bf16x8*)(p + 16);
  return __builtin_shufflevector(lo, hi, 0, 1, 2, 3, 4, 5, 6, 7,
                                 8, 9, 10, 11, 12, 13, 14, 15);
}

// B fragment (32x16 bf16, KxN) from W stored row-major [N][K] (W^T as B).
__device__ inline bf16x16 load_frag_B(const unsigned short* base, int stride, int lane) {
  const unsigned short* p = base + (size_t)(lane & 15) * stride + ((lane >> 4) & 1) * 8;
  bf16x8 lo = *(const bf16x8*)(p);
  bf16x8 hi = *(const bf16x8*)(p + 16);
  return __builtin_shufflevector(lo, hi, 0, 1, 2, 3, 4, 5, 6, 7,
                                 8, 9, 10, 11, 12, 13, 14, 15);
}

__device__ inline f32x8 wmma_bf16(bf16x16 a, bf16x16 b, f32x8 c) {
  return __builtin_amdgcn_wmma_f32_16x16x32_bf16(false, a, false, b,
                                                 (short)0, c, false, false);
}

__device__ inline float sigmoidf_(float x) { return 1.0f / (1.0f + __expf(-x)); }

// ------------------------------- prep kernels ------------------------------

__global__ void convert_bf16_kernel(const float* __restrict__ src,
                                    unsigned short* __restrict__ dst, int n) {
  int i = blockIdx.x * blockDim.x + threadIdx.x;
  if (i < n) dst[i] = f2bf(src[i]);
}

__global__ void bias_kernel(const float* __restrict__ bih,
                            const float* __restrict__ bhh,
                            float* __restrict__ bias) {
  int i = blockIdx.x * blockDim.x + threadIdx.x;
  if (i < GDIM) bias[i] = bih[i] + bhh[i];
}

__global__ void embed_kernel(const int* __restrict__ tgt,
                             const float* __restrict__ emb,
                             unsigned short* __restrict__ Xbf) {
  const long long total = (long long)TPRIME * BATCH * HIDN;
  for (long long e = blockIdx.x * (long long)blockDim.x + threadIdx.x; e < total;
       e += (long long)gridDim.x * blockDim.x) {
    int h = (int)(e % HIDN);
    int b = (int)((e / HIDN) % BATCH);
    int t = (int)(e / ((long long)HIDN * BATCH));
    int idx = tgt[b * 128 + t];                                 // tgt: [B, 128]
    float v = (idx == 0) ? 0.0f : emb[(size_t)idx * HIDN + h];  // padding_idx=0
    Xbf[e] = f2bf(v);                                           // row = t*64+b
  }
}

__global__ void init_state_kernel(const float* __restrict__ h0,
                                  const float* __restrict__ c0,
                                  unsigned short* __restrict__ hbf,
                                  float* __restrict__ cws,
                                  unsigned* __restrict__ cnt) {
  int i = blockIdx.x * blockDim.x + threadIdx.x;
  if (i < BATCH * HIDN) {
    hbf[i] = f2bf(h0[i]);  // ping buffer 0
    cws[i] = c0[i];
  }
  if (i == 0) *cnt = 0u;   // reset persistent-grid barrier every launch
}

// --------------------------- input GEMM (parallel) -------------------------
// WG = 8 waves, tile 64(M) x 256(N). Wave w owns strips n0=bx*256+w*32 and
// n0+16. A (64x32) double-buffered in LDS; async copy overlaps WMMA.
__global__ __launch_bounds__(256) void gemm_in_kernel(
    const unsigned short* __restrict__ Xbf, const unsigned short* __restrict__ Wbf,
    const float* __restrict__ bias, float* __restrict__ Gout) {
  __shared__ unsigned short As[2][64 * 32];
  const int tid = threadIdx.x;
  const int lane = tid & 31;
  const int wave = tid >> 5;
  const int m0 = blockIdx.y * 64;
  const int n0 = blockIdx.x * 256 + wave * 32;

  // one b128 async copy per thread per chunk: 256 thr x 8 bf16 = 64x32 tile
  const int srow = tid >> 2;
  const int scol = (tid & 3) * 8;
  const unsigned short* sbase = Xbf + (size_t)(m0 + srow) * HIDN + scol;

  f32x8 acc[2][4] = {};
  copy_g2l_b128(sbase, As[0] + srow * 32 + scol);  // prime chunk 0
  for (int kc = 0; kc < 32; ++kc) {
    const int k0 = kc * 32;
    const int buf = kc & 1;
    if (kc + 1 < 32) {  // overlap: issue next chunk, wait only for current
      copy_g2l_b128(sbase + (k0 + 32), As[buf ^ 1] + srow * 32 + scol);
      ASYNC_WAIT(1);    // ASYNCcnt retires in order: oldest (chunk kc) landed
    } else {
      ASYNC_WAIT(0);
    }
    __syncthreads();    // chunk kc visible WG-wide

    bf16x16 b0 = load_frag_B(Wbf + (size_t)n0 * HIDN + k0, HIDN, lane);
    bf16x16 b1 = load_frag_B(Wbf + (size_t)(n0 + 16) * HIDN + k0, HIDN, lane);
#pragma unroll
    for (int mt = 0; mt < 4; ++mt) {  // A fragment reused across 2 B strips
      bf16x16 afrag = load_frag_A(As[buf] + (mt * 16) * 32, 32, lane);
      acc[0][mt] = wmma_bf16(afrag, b0, acc[0][mt]);
      acc[1][mt] = wmma_bf16(afrag, b1, acc[1][mt]);
    }
    __syncthreads();    // consumers done before this buffer is re-staged
  }

  const int rh = ((lane >> 4) & 1) * 8;
#pragma unroll
  for (int s = 0; s < 2; ++s) {
    const int nc = n0 + s * 16 + (lane & 15);
    const float bv = bias[nc];
#pragma unroll
    for (int mt = 0; mt < 4; ++mt)
#pragma unroll
      for (int r = 0; r < 8; ++r) {
        int row = m0 + mt * 16 + rh + r;
        Gout[(size_t)row * GDIM + nc] = acc[s][mt][r] + bv;
      }
  }
}

// ------------------------- recurrent LSTM (persistent) ---------------------
// Dynamic LDS layout (272KB of the WGP's 320KB):
//   [0      , 131072): Wlds  64x1024 bf16  (this WG's W_hh rows, resident)
//   [131072 , 262144): Hlds  64x1024 bf16  (h, restaged per step)
//   [262144 , 278528): lg    4x64x16 f32   (gate exchange)
__global__ __launch_bounds__(128) void lstm_kernel(
    const unsigned short* __restrict__ Whh, const float* __restrict__ Gin,
    unsigned short* __restrict__ hbf /* 2 x [64*1024] ping-pong */,
    float* __restrict__ cws, float* __restrict__ out, unsigned* __restrict__ cnt) {
  extern __shared__ char smem[];
  unsigned short* Wlds = (unsigned short*)smem;
  unsigned short* Hlds = (unsigned short*)(smem + 131072);
  float* lg = (float*)(smem + 262144);  // lg[gate][row][col16]

  const int tid = threadIdx.x;
  const int lane = tid & 31;
  const int gate = tid >> 5;        // wave -> PyTorch gate order i,f,g,o
  const int j0 = blockIdx.x * 16;   // hidden columns owned by this WG

  // ---- one-time W_hh tile preload: local row r = gate*16 + col ----
  {
    int r = tid >> 1;                 // 0..63
    int kh = (tid & 1) * 512;         // half of the 1024-wide K row
    int grow = (r >> 4) * HIDN + j0 + (r & 15);
    const unsigned short* src = Whh + (size_t)grow * HIDN + kh;
    unsigned short* dst = Wlds + r * HIDN + kh;
#pragma unroll 4
    for (int k = 0; k < 512; k += 8) copy_g2l_b128(src + k, dst + k);
  }
  async_join();

  for (int t = 0; t < TPRIME; ++t) {
    const unsigned short* hsrc = hbf + (size_t)(t & 1) * (BATCH * HIDN);
    unsigned short* hdst = hbf + (size_t)((t + 1) & 1) * (BATCH * HIDN);

    // ---- stage h[64x1024] bf16 into LDS (async) ----
    {
      int r = tid >> 1;
      int kh = (tid & 1) * 512;
      const unsigned short* src = hsrc + (size_t)r * HIDN + kh;
      unsigned short* dst = Hlds + r * HIDN + kh;
#pragma unroll 4
      for (int k = 0; k < 512; k += 8) copy_g2l_b128(src + k, dst + k);
    }

    // prefetch this step's G slice (HBM-streamed) while staging h
    const float* gt = Gin + (size_t)t * BATCH * GDIM;
    const int nc = lane & 15;
    const int gcol = gate * HIDN + j0 + nc;
    const int rh = ((lane >> 4) & 1) * 8;
#pragma unroll
    for (int mt = 0; mt < 4; ++mt)
      __builtin_prefetch(gt + (size_t)(mt * 16 + rh) * GDIM + gcol, 0, 1);

    async_join();

    // ---- gates(rec) = h @ W_hh^T : all operands in LDS ----
    f32x8 acc[4] = {};
    for (int k0 = 0; k0 < HIDN; k0 += 32) {
      bf16x16 bfrag = load_frag_B(Wlds + (gate * 16) * HIDN + k0, HIDN, lane);
#pragma unroll
      for (int mt = 0; mt < 4; ++mt) {
        bf16x16 afrag = load_frag_A(Hlds + (mt * 16) * HIDN + k0, HIDN, lane);
        acc[mt] = wmma_bf16(afrag, bfrag, acc[mt]);
      }
    }

    // ---- add precomputed input projection, exchange via LDS ----
#pragma unroll
    for (int mt = 0; mt < 4; ++mt)
#pragma unroll
      for (int r = 0; r < 8; ++r) {
        int row = mt * 16 + rh + r;
        lg[(gate * BATCH + row) * 16 + nc] = acc[mt][r] + gt[(size_t)row * GDIM + gcol];
      }
    __syncthreads();

    // ---- f32 cell update: WG owns cols j0..j0+15 exclusively ----
    for (int e = tid; e < BATCH * 16; e += 128) {
      int row = e >> 4, col = e & 15;
      float iv = lg[(0 * BATCH + row) * 16 + col];
      float fv = lg[(1 * BATCH + row) * 16 + col];
      float gv = lg[(2 * BATCH + row) * 16 + col];
      float ov = lg[(3 * BATCH + row) * 16 + col];
      size_t cidx = (size_t)row * HIDN + j0 + col;
      float cn = sigmoidf_(fv) * cws[cidx] + sigmoidf_(iv) * tanhf(gv);
      cws[cidx] = cn;
      float hn = sigmoidf_(ov) * tanhf(cn);
      hdst[cidx] = f2bf(hn);
      out[((size_t)row * TPRIME + t) * HIDN + j0 + col] = hn;  // outputs[b,t,h]
      if (t == TPRIME - 1) {
        size_t base = (size_t)BATCH * TPRIME * HIDN;
        out[base + cidx] = hn;                         // final h
        out[base + (size_t)BATCH * HIDN + cidx] = cn;  // final c
      }
    }
    __syncthreads();

    if (t < TPRIME - 1) {  // device-scope barrier; all 64 WGs co-resident
      __threadfence();
      if (tid == 0) {
        __hip_atomic_fetch_add(cnt, 1u, __ATOMIC_ACQ_REL, __HIP_MEMORY_SCOPE_AGENT);
        unsigned want = (unsigned)(t + 1) * (unsigned)gridDim.x;
        while (__hip_atomic_load(cnt, __ATOMIC_ACQUIRE, __HIP_MEMORY_SCOPE_AGENT) < want)
          __builtin_amdgcn_s_sleep(2);
      }
      __syncthreads();
      __threadfence();
    }
  }
}

// --------------------------------- launcher --------------------------------

extern "C" void kernel_launch(void* const* d_in, const int* in_sizes, int n_in,
                              void* d_out, int out_size, void* d_ws, size_t ws_size,
                              hipStream_t stream) {
  (void)in_sizes; (void)n_in; (void)out_size; (void)ws_size;
  const int*   tgt = (const int*)d_in[0];
  const float* h0  = (const float*)d_in[1];
  const float* c0  = (const float*)d_in[2];
  // d_in[3] encoder_outputs, d_in[4] src_lengths: unused (attn=None)
  const float* emb = (const float*)d_in[5];
  const float* wih = (const float*)d_in[6];
  const float* whh = (const float*)d_in[7];
  const float* bih = (const float*)d_in[8];
  const float* bhh = (const float*)d_in[9];
  float* out = (float*)d_out;

  char* ws = (char*)d_ws;
  size_t off = 0;
  auto carve = [&](size_t bytes) {
    size_t o = off;
    off = (off + bytes + 255) & ~(size_t)255;
    return o;
  };
  float*          G    = (float*)(ws + carve((size_t)TPRIME * BATCH * GDIM * 4));
  unsigned short* Xbf  = (unsigned short*)(ws + carve((size_t)TPRIME * BATCH * HIDN * 2));
  unsigned short* Wihb = (unsigned short*)(ws + carve((size_t)GDIM * HIDN * 2));
  unsigned short* Whhb = (unsigned short*)(ws + carve((size_t)GDIM * HIDN * 2));
  float*          bias = (float*)(ws + carve((size_t)GDIM * 4));
  unsigned short* hbf  = (unsigned short*)(ws + carve((size_t)2 * BATCH * HIDN * 2));
  float*          cws  = (float*)(ws + carve((size_t)BATCH * HIDN * 4));
  unsigned*       cnt  = (unsigned*)(ws + carve(sizeof(unsigned)));

  const int nW = GDIM * HIDN;
  convert_bf16_kernel<<<(nW + 255) / 256, 256, 0, stream>>>(wih, Wihb, nW);
  convert_bf16_kernel<<<(nW + 255) / 256, 256, 0, stream>>>(whh, Whhb, nW);
  bias_kernel<<<(GDIM + 255) / 256, 256, 0, stream>>>(bih, bhh, bias);
  embed_kernel<<<8192, 256, 0, stream>>>(tgt, emb, Xbf);
  init_state_kernel<<<(BATCH * HIDN + 255) / 256, 256, 0, stream>>>(h0, c0, hbf, cws, cnt);

  // 16 N-tiles x 127 M-tiles, 8 waves each, double-buffered async staging.
  gemm_in_kernel<<<dim3(16, TPRIME), 256, 0, stream>>>(Xbf, Wihb, bias, G);

  // 272KB dynamic LDS: W_hh tile resident + h staging + gate exchange.
  lstm_kernel<<<LSTM_WGS, 128, 278528, stream>>>(Whhb, G, hbf, cws, out, cnt);
}
